// Decoder_6270652252197
// MI455X (gfx1250) — compile-verified, matrix operand order checked by reading, not run on hardware
//
#include <hip/hip_runtime.h>
#include <hip/hip_bf16.h>
#include <math.h>

typedef __attribute__((ext_vector_type(16))) __bf16 v16bf;
typedef __attribute__((ext_vector_type(8)))  float  v8f;

namespace {

constexpr int Bdim = 32;
constexpr int Tdim = 1024;

union V16U { v16bf v; float4 f[2]; };

// Load one 16x16x32 bf16 WMMA fragment half-pair for this lane.
// Layout (CDNA5 ISA, 16-bit A 16x32): lanes 0-15 hold K = k0..k0+7 and
// k0+16..k0+23; lanes 16-31 hold K = k0+8.. and k0+24.. (caller passes the
// pointer already offset by kb + (lane&16 ? 8 : 0)). Two b128 loads.
__device__ inline v16bf load_frag(const __bf16* p) {
  V16U u;
  u.f[0] = *(const float4*)(p);
  u.f[1] = *(const float4*)(p + 16);
  return u.v;
}

__device__ inline float sigmoidf_(float x) {
  return 1.0f / (1.0f + __expf(-x));
}

// src f32 [K][N] row-major -> dst bf16 [N][K] (transposed, WMMA-B friendly)
__global__ void __launch_bounds__(256)
k_convert_transpose(const float* __restrict__ src, __bf16* __restrict__ dst,
                    int K, int N) {
  int j = blockIdx.x * 256 + threadIdx.x;
  if (j < K * N) {
    int k = j / N, n = j % N;
    dst[(size_t)n * K + k] = (__bf16)src[j];
  }
}

__global__ void __launch_bounds__(256)
k_convert_flat(const float* __restrict__ src, __bf16* __restrict__ dst, int n) {
  int j = blockIdx.x * 256 + threadIdx.x;
  if (j < n) dst[j] = (__bf16)src[j];
}

// C[m,n] = act( sum_k A[m,k]*Bt[n,k] + bias[n] )
// Block = 8 waves, one 16-row M tile per block; wave w owns N tiles w, w+8, ...
// NTPW=6 -> N=768 (GRU input projections), NTPW=2 -> N=256 (output dense).
// mode 0: store xp[T][B][N], A row = b*T+t        (layer-1, A = decoder_input)
// mode 1: store xp[T][B][N], A row = t*B+b        (layer-2, A = p1)
// mode 2: recon[b][T-1-t][n] = tanhf(v)  (f32)    (projection, A = p2)
template<int NTPW>
__global__ void __launch_bounds__(256)
k_gemm(const __bf16* __restrict__ A, int lda,
       const __bf16* __restrict__ Bt,     // [N][K] bf16 (pre-transposed)
       const float* __restrict__ bias,    // [N] f32
       int K,
       void* __restrict__ out,
       int mode, int flip)
{
  const int lane  = threadIdx.x & 31;
  const int wave  = threadIdx.x >> 5;
  const int mtile = blockIdx.x;
  const int rsel  = lane & 15;
  const int k0sel = (lane & 16) ? 8 : 0;
  const int N     = NTPW * 8 * 16;

  v8f acc[NTPW];
#pragma unroll
  for (int i = 0; i < NTPW; i++)
#pragma unroll
    for (int g = 0; g < 8; g++) acc[i][g] = 0.0f;

  const __bf16* Arow = A + (size_t)(mtile * 16 + rsel) * lda;
  for (int kb = 0; kb < K; kb += 32) {
    v16bf af = load_frag(Arow + kb + k0sel);
#pragma unroll
    for (int i = 0; i < NTPW; i++) {
      const int ntile = wave + i * 8;
      const __bf16* Brow = Bt + (size_t)(ntile * 16 + rsel) * K + kb + k0sel;
      v16bf bf = load_frag(Brow);
      acc[i] = __builtin_amdgcn_wmma_f32_16x16x32_bf16(
          false, af, false, bf, (short)0, acc[i], false, false);
    }
  }

  // C/D layout: VGPR g: lanes 0-15 -> (M=g, N=lane); lanes 16-31 -> (M=g+8, N=lane-16)
#pragma unroll
  for (int i = 0; i < NTPW; i++) {
    const int col = (wave + i * 8) * 16 + rsel;
    const float bv = bias[col];
#pragma unroll
    for (int g = 0; g < 8; g++) {
      const int gm = mtile * 16 + g + ((lane & 16) ? 8 : 0);
      float v = acc[i][g] + bv;
      if (mode == 2) {
        int t = gm / Bdim, b = gm % Bdim;
        int td = Tdim - 1 - t;      // recon time flip
        ((float*)out)[((size_t)b * Tdim + td) * N + col] = tanhf(v);
      } else {
        int t, b;
        if (mode == 0) { b = gm / Tdim; t = gm % Tdim; }
        else           { t = gm / Bdim; b = gm % Bdim; }
        if (flip) t = Tdim - 1 - t; // backward dir: xp stored in scan order
        ((__bf16*)out)[((size_t)t * Bdim + b) * N + col] = (__bf16)v;
      }
    }
  }
}

// Recurrent GRU scan. grid = (2 dirs, 2 batch halves), 256 threads (8 waves).
// Each block owns 16 independent batch rows; h kept in LDS bf16, rec scratch
// in LDS f32. Per step: 48 tiles (1x48 over N=768), wave w -> tiles w+8i,
// 8 K-chunks each => 48 WMMAs/wave/step; then fused gate math. The compiler
// hoists the loop-invariant U^T fragments into (extended) VGPRs, so the
// per-step critical path is ds_load of h + WMMA chain + gates. We prefetch
// the next timestep's xp slab (24 KB) during the WMMA phase so the gate
// phase's global reads hit cache.
__global__ void __launch_bounds__(256)
k_gru_scan(const __bf16* __restrict__ xpF, const __bf16* __restrict__ xpB, // [T][32][768]
           const __bf16* __restrict__ UtF, const __bf16* __restrict__ UtB, // [768][256]
           const float* __restrict__ bF, const float* __restrict__ bB,     // [2][768]
           const float* __restrict__ h0,                                   // [32][256]
           __bf16* __restrict__ outF, __bf16* __restrict__ outB)           // [T][32][512] (+dir col off)
{
  __shared__ __align__(16) __bf16 hS[16 * 256];   // 8 KB
  __shared__ float recS[16 * 768];                // 48 KB

  const int dir  = blockIdx.x;
  const int half = blockIdx.y;
  const __bf16* xp  = dir ? xpB : xpF;
  const __bf16* Ut  = dir ? UtB : UtF;
  const float* brec = (dir ? bB : bF) + 768;      // b[1] (recurrent bias)
  __bf16* out = dir ? outB : outF;
  const int b0 = half * 16;

  const int tid   = threadIdx.x;
  const int lane  = tid & 31;
  const int wave  = tid >> 5;
  const int rsel  = lane & 15;
  const int k0sel = (lane & 16) ? 8 : 0;
  const int rrow  = (lane & 16) ? 8 : 0;

  for (int j = tid; j < 16 * 256; j += 256)
    hS[j] = (__bf16)h0[(b0 + (j >> 8)) * 256 + (j & 255)];
  __syncthreads();

  for (int t = 0; t < Tdim; t++) {
    // prefetch next step's xp slab: 16 rows * 768 bf16 = 24576 B = 192 lines
    if (t + 1 < Tdim && tid < 192) {
      const char* nxt = (const char*)(xp + ((size_t)(t + 1) * Bdim + b0) * 768);
      __builtin_prefetch(nxt + (size_t)tid * 128, 0, 0);  // global_prefetch_b8
    }

    // rec = h @ U (bias added in gate phase); U^T fragments live in VGPRs
#pragma unroll
    for (int i = 0; i < 6; i++) {
      const int ntile = wave + i * 8;
      v8f acc;
#pragma unroll
      for (int g = 0; g < 8; g++) acc[g] = 0.0f;
#pragma unroll
      for (int kb = 0; kb < 256; kb += 32) {
        v16bf a  = load_frag(&hS[rsel * 256 + kb + k0sel]);
        v16bf bm = load_frag(Ut + (size_t)(ntile * 16 + rsel) * 256 + kb + k0sel);
        acc = __builtin_amdgcn_wmma_f32_16x16x32_bf16(
            false, a, false, bm, (short)0, acc, false, false);
      }
      const int col = ntile * 16 + rsel;
#pragma unroll
      for (int g = 0; g < 8; g++)
        recS[(rrow + g) * 768 + col] = acc[g];
    }
    __syncthreads();

    // gate phase: thread tid owns feature column c=tid for all 16 batch rows
    const __bf16* xpt  = xp  + ((size_t)t * Bdim + b0) * 768;
    __bf16*       outt = out + ((size_t)t * Bdim + b0) * 512;
    const int c = tid;
    for (int bl = 0; bl < 16; bl++) {
      float xz = (float)xpt[bl * 768 + c];
      float xr = (float)xpt[bl * 768 + 256 + c];
      float xh = (float)xpt[bl * 768 + 512 + c];
      float rz = recS[bl * 768 + c]       + brec[c];
      float rr = recS[bl * 768 + 256 + c] + brec[256 + c];
      float rh = recS[bl * 768 + 512 + c] + brec[512 + c];
      float hp = (float)hS[bl * 256 + c];
      float z  = sigmoidf_(xz + rz);
      float r  = sigmoidf_(xr + rr);
      float hh = tanhf(xh + r * rh);
      float hn = z * hp + (1.0f - z) * hh;
      __bf16 hb = (__bf16)hn;
      hS[bl * 256 + c] = hb;
      outt[bl * 512 + c] = hb;
    }
    __syncthreads();
  }
}

} // namespace

extern "C" void kernel_launch(void* const* d_in, const int* in_sizes, int n_in,
                              void* d_out, int out_size, void* d_ws, size_t ws_size,
                              hipStream_t stream) {
  (void)in_sizes; (void)n_in; (void)out_size; (void)ws_size;
  char* ws = (char*)d_ws;

  const size_t SZ_U  = (size_t)768 * 256 * 2;  // bf16 [768][256]
  const size_t SZ_W2 = (size_t)768 * 512 * 2;  // bf16 [768][512]
  const size_t SZ_WP = (size_t)256 * 512 * 2;  // bf16 [256][512]
  size_t o = 0;
  const size_t o_w1f = o; o += SZ_U;
  const size_t o_w1b = o; o += SZ_U;
  const size_t o_u1f = o; o += SZ_U;
  const size_t o_u1b = o; o += SZ_U;
  const size_t o_u2f = o; o += SZ_U;
  const size_t o_u2b = o; o += SZ_U;
  const size_t o_w2f = o; o += SZ_W2;
  const size_t o_w2b = o; o += SZ_W2;
  const size_t o_wp  = o; o += SZ_WP;
  const size_t o_xbf = o; o += (size_t)32768 * 256 * 2;  // x as bf16
  const size_t o_xpf = o; o += (size_t)32768 * 768 * 2;  // xp fwd (reused by layer 2)
  const size_t o_xpb = o; o += (size_t)32768 * 768 * 2;  // xp bwd (reused by layer 2)
  const size_t o_p1  = o; o += (size_t)32768 * 512 * 2;  // [T][B][512] bf16
  const size_t o_p2  = o; o += (size_t)32768 * 512 * 2;  // [T][B][512] bf16

  const float* enc_h = (const float*)d_in[0];
  const float* x     = (const float*)d_in[1];

  auto ct = [&](int idx, size_t off, int K, int N) {
    int n = K * N;
    k_convert_transpose<<<(n + 255) / 256, 256, 0, stream>>>(
        (const float*)d_in[idx], (__bf16*)(ws + off), K, N);
  };
  ct(2,  o_w1f, 256, 768);   // W1f
  ct(5,  o_w1b, 256, 768);   // W1b
  ct(3,  o_u1f, 256, 768);   // U1f
  ct(6,  o_u1b, 256, 768);   // U1b
  ct(9,  o_u2f, 256, 768);   // U2f
  ct(12, o_u2b, 256, 768);   // U2b
  ct(8,  o_w2f, 512, 768);   // W2f
  ct(11, o_w2b, 512, 768);   // W2b
  ct(14, o_wp,  512, 256);   // Wp

  {
    int n = 32768 * 256;
    k_convert_flat<<<(n + 255) / 256, 256, 0, stream>>>(x, (__bf16*)(ws + o_xbf), n);
  }

  // layer-1 input projections: xp = x @ W + b[0]  (bwd: time-flipped store)
  k_gemm<6><<<2048, 256, 0, stream>>>((const __bf16*)(ws + o_xbf), 256,
      (const __bf16*)(ws + o_w1f), (const float*)d_in[4], 256, ws + o_xpf, 0, 0);
  k_gemm<6><<<2048, 256, 0, stream>>>((const __bf16*)(ws + o_xbf), 256,
      (const __bf16*)(ws + o_w1b), (const float*)d_in[7], 256, ws + o_xpb, 0, 1);

  // layer-1 recurrent scans (fwd + bwd, 2 batch halves each)
  dim3 sg(2, 2);
  k_gru_scan<<<sg, 256, 0, stream>>>(
      (const __bf16*)(ws + o_xpf), (const __bf16*)(ws + o_xpb),
      (const __bf16*)(ws + o_u1f), (const __bf16*)(ws + o_u1b),
      (const float*)d_in[4], (const float*)d_in[7], enc_h,
      (__bf16*)(ws + o_p1), (__bf16*)(ws + o_p1) + 256);

  // layer-2 input projections from p1 [T][B][512]
  k_gemm<6><<<2048, 256, 0, stream>>>((const __bf16*)(ws + o_p1), 512,
      (const __bf16*)(ws + o_w2f), (const float*)d_in[10], 512, ws + o_xpf, 1, 0);
  k_gemm<6><<<2048, 256, 0, stream>>>((const __bf16*)(ws + o_p1), 512,
      (const __bf16*)(ws + o_w2b), (const float*)d_in[13], 512, ws + o_xpb, 1, 1);

  // layer-2 recurrent scans
  k_gru_scan<<<sg, 256, 0, stream>>>(
      (const __bf16*)(ws + o_xpf), (const __bf16*)(ws + o_xpb),
      (const __bf16*)(ws + o_u2f), (const __bf16*)(ws + o_u2b),
      (const float*)d_in[10], (const float*)d_in[13], enc_h,
      (__bf16*)(ws + o_p2), (__bf16*)(ws + o_p2) + 256);

  // projection: recon[b][T-1-t] = tanh(p2[t] @ Wp + bp), f32 out
  k_gemm<2><<<2048, 256, 0, stream>>>((const __bf16*)(ws + o_p2), 512,
      (const __bf16*)(ws + o_wp), (const float*)d_in[15], 512, d_out, 2, 0);
}
